// LSTMLayer_57432302682693
// MI455X (gfx1250) — matrix operand order repaired and checked by reading, not compiled, not run on hardware
//
#include <hip/hip_runtime.h>

// LSTM cell:  gates = [hidden|inputs] @ [Wf|Wi|Wo|Wc]^T + b  (bf16 WMMA, f32 acc)
//             c' = sig(f)*c + sig(i)*tanh(chat) ; h' = sig(o)*tanh(c')
// B=8192, IN=1024, H=1024, K=2048.
//
// Fast path (ws >= 48MB): pass 1 converts X=[hidden|inputs] and W to bf16 in ws;
// pass 2 is a conversion-free WMMA GEMM with fused LSTM epilogue, using
// 128-thread blocks (4 waves): in CU mode a 4-wave workgroup = 2 waves/SIMD,
// giving a 256-VGPR budget so fragments+staging never spill.

typedef __attribute__((ext_vector_type(16))) __bf16 bvec16;
typedef __attribute__((ext_vector_type(8)))  __bf16 bvec8;
typedef __attribute__((ext_vector_type(4)))  __bf16 bvec4;
typedef __attribute__((ext_vector_type(4)))  float  fvec4;
typedef __attribute__((ext_vector_type(8)))  float  v8f;

union BV { bvec16 v; bvec8 h[2]; };

__device__ __forceinline__ bvec4 cvt_bf16x4(float4 v) {
  fvec4 f; f.x = v.x; f.y = v.y; f.z = v.z; f.w = v.w;
  return __builtin_convertvector(f, bvec4);   // v_cvt_pk_bf16_f32 x2
}

__device__ __forceinline__ float sigmoid_f(float x) {
  return 1.0f / (1.0f + __expf(-x));
}
__device__ __forceinline__ float tanh_f(float x) {
  x = fminf(fmaxf(x, -15.0f), 15.0f);
  float e = __expf(-2.0f * x);
  return (1.0f - e) / (1.0f + e);
}

#define BK   32    // K step (bf16 wmma depth)
#define SROW 80    // LDS row stride bytes: 64B data + 16B pad (b128-aligned, conflict-free)

// bf16 fast-path block: 64 M-rows x 32 h-cols x 4 gates, 128 threads = 4 waves
#define BMP    64
#define BHP    32
#define APART  (BMP * SROW)                 // 5120
#define BPART  (4 * BHP * SROW)             // 10240
#define BUFB   (APART + BPART)              // 15360

// fallback block: 128 M-rows x 32 h-cols x 4 gates, 256 threads = 8 waves
#define BM   128
#define BH   32
#define ABYTES (BM * SROW)                  // 10240
#define BBYTES (4 * BH * SROW)              // 10240

#define NBATCH 8192
#define NH     1024
#define NK_DIM 2048
#define XBF_ELEMS ((size_t)NBATCH * NK_DIM)          // 16.7M bf16 = 32MB
#define WBF_ELEMS ((size_t)4 * NH * NK_DIM)          // 8.4M bf16 = 16MB
#define WS_NEED   (2 * (XBF_ELEMS + WBF_ELEMS))      // bytes (bf16 = 2B)

// ---------------------------------------------------------------- pass 1: cvt
__global__ __launch_bounds__(256)
void cvt_pack_kernel(const float* __restrict__ inputs,
                     const float* __restrict__ hidden,
                     const float* __restrict__ Wf, const float* __restrict__ Wi,
                     const float* __restrict__ Wo, const float* __restrict__ Wc,
                     __bf16* __restrict__ xbf, __bf16* __restrict__ wbf)
{
  constexpr int NX4 = (int)(XBF_ELEMS / 4);   // 4,194,304 float4s of X
  constexpr int NW4 = (int)(WBF_ELEMS / 4);   // 2,097,152 float4s of W
  int idx = blockIdx.x * 256 + threadIdx.x;
  if (idx < NX4) {
    int b  = idx >> 9;              // 512 float4 per 2048-col row
    int k4 = idx & 511;
    int k  = k4 * 4;
    const float* src = (k < NH) ? (hidden + b * NH + k) : (inputs + b * NH + (k - NH));
    float4 v = *(const float4*)src;
    *(bvec4*)(xbf + (size_t)b * NK_DIM + k) = cvt_bf16x4(v);
  } else if (idx < NX4 + NW4) {
    int j   = idx - NX4;
    int g   = j >> 19;              // 1024*512 float4 per gate matrix
    int rem = j & ((1 << 19) - 1);
    int row = rem >> 9;
    int k   = (rem & 511) * 4;
    const float* W[4] = { Wf, Wi, Wo, Wc };
    float4 v = *(const float4*)(W[g] + (size_t)row * NK_DIM + k);
    *(bvec4*)(wbf + ((size_t)(g * NH + row)) * NK_DIM + k) = cvt_bf16x4(v);
  }
}

// ------------------------------------------------- pass 2: GEMM, bf16 sources
// 4 waves = (mpair 0..1, nt 0..1); each wave: 2 M-tiles x 4 gates
// -> 8 WMMAs per k-step, 12 ds_load_b128, accumulators register-resident.
__global__ __launch_bounds__(128)
__attribute__((amdgpu_waves_per_eu(2)))
void lstm_gemm_bf16_kernel(const __bf16* __restrict__ xbf,
                           const __bf16* __restrict__ wbf,
                           const float* __restrict__ cin,
                           const float* __restrict__ bfp, const float* __restrict__ bip,
                           const float* __restrict__ bop, const float* __restrict__ bcp,
                           float* __restrict__ hout, float* __restrict__ cout)
{
  constexpr int K  = NK_DIM;
  constexpr int NK = K / BK;       // 64 k-steps

  __shared__ unsigned char lds[2 * BUFB];   // 30 KB, double buffered

  const int tid   = threadIdx.x;
  const int wave  = tid >> 5;
  const int lane  = tid & 31;
  const int l16   = lane & 15;
  const int hf    = lane >> 4;
  const int mpair = wave >> 1;          // 0..1
  const int nt    = wave & 1;           // 0..1

  const int Mbase = blockIdx.x * BMP;
  const int Hbase = blockIdx.y * BHP;

  // staging (128 threads):
  //  A tile 64x32: 2 threads/row, 16 bf16 (2x b128) each
  //  B tile 4x32x32: 1 thread/row, 32 bf16 (4x b128) each
  const int arow  = tid >> 1;            // 0..63
  const int ahalf = (tid & 1) * 16;      // bf16 col 0 or 16
  const int aoff  = (Mbase + arow) * K + ahalf;                      // + k1
  const int g_    = tid >> 5;            // B: gate 0..3
  const int hrow  = tid & 31;            // B: row within gate tile
  const int boff  = (g_ * NH + Hbase + hrow) * K;                    // + k1

  v8f acc[4][2] = {};
  uint4 areg[2], breg[4];

  auto gload = [&](int k1) {
    const __bf16* xb = xbf + k1;         // uniform base -> saddr + voffset
    areg[0] = *(const uint4*)(xb + aoff);
    areg[1] = *(const uint4*)(xb + aoff + 8);
    const __bf16* wb = wbf + k1;
#pragma unroll
    for (int i = 0; i < 4; ++i)
      breg[i] = *(const uint4*)(wb + boff + 8 * i);
  };

  auto lstore = [&](int buf) {
    unsigned char* base = lds + buf * BUFB;
    unsigned char* adst = base + arow * SROW + ahalf * 2;
    *(uint4*)(adst)      = areg[0];
    *(uint4*)(adst + 16) = areg[1];
    unsigned char* bdst = base + APART + tid * SROW;   // tid == g_*32+hrow
#pragma unroll
    for (int i = 0; i < 4; ++i)
      *(uint4*)(bdst + 16 * i) = breg[i];
  };

  // fragment loads + 8 WMMAs for one staged k-tile
  auto mma_step = [&](int buf) {
    const unsigned char* base = lds + buf * BUFB;
    const unsigned char* bb   = base + APART;
    // A frag (ISA 16-bit A layout): lane row = l16; elems 0..7 = K[hf*8..],
    // elems 8..15 = K[16+hf*8..]
    BV af[2];
#pragma unroll
    for (int mt = 0; mt < 2; ++mt) {
      const unsigned char* ar = base + (mpair * 32 + mt * 16 + l16) * SROW;
      af[mt].h[0] = *(const bvec8*)(ar + hf * 16);
      af[mt].h[1] = *(const bvec8*)(ar + 32 + hf * 16);
    }
    // B frag (ISA 16-bit B layout): lane col = l16; elems 0..7 = K[hf*8..],
    // elems 8..15 = K[16+hf*8..]
    BV bfv[4];
#pragma unroll
    for (int g = 0; g < 4; ++g) {
      const unsigned char* br = bb + (g * BHP + nt * 16 + l16) * SROW + hf * 16;
      bfv[g].h[0] = *(const bvec8*)(br);
      bfv[g].h[1] = *(const bvec8*)(br + 32);
    }
#pragma unroll
    for (int g = 0; g < 4; ++g)
#pragma unroll
      for (int mt = 0; mt < 2; ++mt)
        acc[g][mt] = __builtin_amdgcn_wmma_f32_16x16x32_bf16(
            false, af[mt].v, false, bfv[g].v, (short)0, acc[g][mt], false, false);
  };

  gload(0);
  lstore(0);
  __syncthreads();

#pragma unroll 1
  for (int kt = 0; kt < NK - 1; ++kt) {     // branch-free steady state
    const int buf = kt & 1;
    gload((kt + 1) * BK);
    mma_step(buf);
    lstore(buf ^ 1);
    __syncthreads();
  }
  mma_step((NK - 1) & 1);                   // peeled tail

  // fused LSTM epilogue; C/D layout: reg r -> m = r + 8*hf, n = l16
  const int h = Hbase + nt * 16 + l16;
  const float vbf = bfp[h], vbi = bip[h], vbo = bop[h], vbc = bcp[h];
#pragma unroll
  for (int mt = 0; mt < 2; ++mt) {
    const int mrow = Mbase + mpair * 32 + mt * 16 + 8 * hf;
#pragma unroll
    for (int r = 0; r < 8; ++r) {
      const size_t idx = (size_t)(mrow + r) * NH + h;
      const float fg = sigmoid_f(acc[0][mt][r] + vbf);
      const float ig = sigmoid_f(acc[1][mt][r] + vbi);
      const float og = sigmoid_f(acc[2][mt][r] + vbo);
      const float cg = tanh_f  (acc[3][mt][r] + vbc);
      const float cn = fg * cin[idx] + ig * cg;
      hout[idx] = og * tanh_f(cn);
      cout[idx] = cn;
    }
  }
}

// ------------------------------------------------ fallback: fused f32 sources
__global__ __launch_bounds__(256)
__attribute__((amdgpu_waves_per_eu(2)))
void lstm_fused_kernel(const float* __restrict__ inputs,
                       const float* __restrict__ hidden,
                       const float* __restrict__ cin,
                       const float* __restrict__ Wf, const float* __restrict__ bfp,
                       const float* __restrict__ Wi, const float* __restrict__ bip,
                       const float* __restrict__ Wo, const float* __restrict__ bop,
                       const float* __restrict__ Wc, const float* __restrict__ bcp,
                       float* __restrict__ hout, float* __restrict__ cout)
{
  constexpr int H = NH;
  constexpr int K = NK_DIM;
  constexpr int NK = K / BK;

  __shared__ unsigned char lds[2 * (ABYTES + BBYTES)];

  const int tid   = threadIdx.x;
  const int wave  = tid >> 5;
  const int lane  = tid & 31;
  const int l16   = lane & 15;
  const int hf    = lane >> 4;
  const int mpair = wave >> 1;
  const int nt    = wave & 1;

  const int Mbase = blockIdx.x * BM;
  const int Hbase = blockIdx.y * BH;

  const float* W[4] = { Wf, Wi, Wo, Wc };

  const int lrow = tid >> 3;
  const int lcol = (tid & 7) * 4;
  const int aoff = (Mbase + lrow) * 1024 + lcol;
  const int boff = lrow * K + lcol;

  v8f acc[4][2] = {};
  float4 areg[4], breg[4];

  auto gload = [&](int k1) {
    const float* xbase = ((k1 < H) ? hidden : (inputs - H)) + k1;
#pragma unroll
    for (int j = 0; j < 4; ++j)
      areg[j] = *(const float4*)(xbase + aoff + j * (32 * 1024));
#pragma unroll
    for (int g = 0; g < 4; ++g)
      breg[g] = *(const float4*)(W[g] + Hbase * K + k1 + boff);
  };

  auto lstore = [&](int buf) {
    unsigned char* base = lds + buf * (ABYTES + BBYTES);
#pragma unroll
    for (int j = 0; j < 4; ++j)
      *(bvec4*)(base + (lrow + 32 * j) * SROW + lcol * 2) = cvt_bf16x4(areg[j]);
    unsigned char* bb = base + ABYTES;
#pragma unroll
    for (int g = 0; g < 4; ++g)
      *(bvec4*)(bb + (g * BH + lrow) * SROW + lcol * 2) = cvt_bf16x4(breg[g]);
  };

  gload(0);
  lstore(0);
  __syncthreads();

  for (int kt = 0; kt < NK; ++kt) {
    const int buf = kt & 1;
    if (kt + 1 < NK) gload((kt + 1) * BK);

    const unsigned char* base = lds + buf * (ABYTES + BBYTES);
    const unsigned char* bb   = base + ABYTES;

    BV af[2];
#pragma unroll
    for (int mt = 0; mt < 2; ++mt) {
      const unsigned char* ar = base + (mpair * 32 + mt * 16 + l16) * SROW;
      af[mt].h[0] = *(const bvec8*)(ar + hf * 16);
      af[mt].h[1] = *(const bvec8*)(ar + 32 + hf * 16);
    }
    BV bfv[4];
#pragma unroll
    for (int g = 0; g < 4; ++g) {
      const unsigned char* br = bb + (g * BH + nt * 16 + l16) * SROW + hf * 16;
      bfv[g].h[0] = *(const bvec8*)(br);
      bfv[g].h[1] = *(const bvec8*)(br + 32);
    }

#pragma unroll
    for (int g = 0; g < 4; ++g)
#pragma unroll
      for (int mt = 0; mt < 2; ++mt)
        acc[g][mt] = __builtin_amdgcn_wmma_f32_16x16x32_bf16(
            false, af[mt].v, false, bfv[g].v, (short)0, acc[g][mt], false, false);

    if (kt + 1 < NK) lstore(buf ^ 1);
    __syncthreads();
  }

  const int h = Hbase + nt * 16 + l16;
  const float vbf = bfp[h], vbi = bip[h], vbo = bop[h], vbc = bcp[h];
#pragma unroll
  for (int mt = 0; mt < 2; ++mt) {
    const int mrow = Mbase + mpair * 32 + mt * 16 + 8 * hf;
#pragma unroll
    for (int r = 0; r < 8; ++r) {
      const size_t idx = (size_t)(mrow + r) * NH + h;
      const float fg = sigmoid_f(acc[0][mt][r] + vbf);
      const float ig = sigmoid_f(acc[1][mt][r] + vbi);
      const float og = sigmoid_f(acc[2][mt][r] + vbo);
      const float cg = tanh_f  (acc[3][mt][r] + vbc);
      const float cn = fg * cin[idx] + ig * cg;
      hout[idx] = og * tanh_f(cn);
      cout[idx] = cn;
    }
  }
}

extern "C" void kernel_launch(void* const* d_in, const int* in_sizes, int n_in,
                              void* d_out, int out_size, void* d_ws, size_t ws_size,
                              hipStream_t stream) {
  (void)in_sizes; (void)n_in; (void)out_size;
  const float* inputs = (const float*)d_in[0];
  const float* hidden = (const float*)d_in[1];
  const float* cin    = (const float*)d_in[2];
  const float* Wf = (const float*)d_in[3];  const float* bfp = (const float*)d_in[4];
  const float* Wi = (const float*)d_in[5];  const float* bip = (const float*)d_in[6];
  const float* Wo = (const float*)d_in[7];  const float* bop = (const float*)d_in[8];
  const float* Wc = (const float*)d_in[9];  const float* bcp = (const float*)d_in[10];

  float* hout = (float*)d_out;
  float* cout = hout + (size_t)NBATCH * NH;

  if (d_ws != nullptr && ws_size >= WS_NEED) {
    __bf16* xbf = (__bf16*)d_ws;
    __bf16* wbf = xbf + XBF_ELEMS;
    const int n4 = (int)((XBF_ELEMS + WBF_ELEMS) / 4);
    cvt_pack_kernel<<<(n4 + 255) / 256, 256, 0, stream>>>(
        inputs, hidden, Wf, Wi, Wo, Wc, xbf, wbf);
    dim3 grid(NBATCH / BMP, NH / BHP);   // 128 x 32 blocks
    lstm_gemm_bf16_kernel<<<grid, dim3(128), 0, stream>>>(
        xbf, wbf, cin, bfp, bip, bop, bcp, hout, cout);
  } else {
    dim3 grid(NBATCH / BM, NH / BH);     // 64 x 32 blocks
    lstm_fused_kernel<<<grid, dim3(256), 0, stream>>>(
        inputs, hidden, cin, Wf, bfp, Wi, bip, Wo, bop, Wc, bcp, hout, cout);
  }
}
